// Model_26216480374962
// MI455X (gfx1250) — matrix operand order repaired and checked by reading, not compiled, not run on hardware
//
#include <hip/hip_runtime.h>
#include <math.h>

#define BSZ   32
#define NPOS  243
#define DIM   544
#define NH    8
#define HD    68
#define NSEED 17
#define UPD   1088      // UP*D
#define MLP1N 2176
#define MLP2N 1088
#define MROWS (BSZ*NPOS)   // 7776 (multiple of 32)
#define BTW   (2*NPOS-1)   // 485

typedef __attribute__((ext_vector_type(16))) __bf16 v16bf;
typedef __attribute__((ext_vector_type(8)))  float  v8f;

__device__ __forceinline__ float geluf(float x) {
  return 0.5f * x * (1.0f + erff(x * 0.70710678118654752f));
}

// ---------------- LayerNorm (fp32, one block per row) ----------------
__global__ __launch_bounds__(256) void ln_kernel(const float* __restrict__ X,
    const float* __restrict__ g, const float* __restrict__ b,
    float* __restrict__ Y) {
  __shared__ float r1[256], r2[256];
  const int row = blockIdx.x;
  const float* x = X + (long)row * DIM;
  float s = 0.f, sq = 0.f;
  for (int c = threadIdx.x; c < DIM; c += 256) { float v = x[c]; s += v; sq += v * v; }
  r1[threadIdx.x] = s; r2[threadIdx.x] = sq; __syncthreads();
  for (int k = 128; k > 0; k >>= 1) {
    if (threadIdx.x < (unsigned)k) { r1[threadIdx.x] += r1[threadIdx.x + k]; r2[threadIdx.x] += r2[threadIdx.x + k]; }
    __syncthreads();
  }
  const float mean = r1[0] / (float)DIM;
  const float var  = r2[0] / (float)DIM - mean * mean;
  const float inv  = rsqrtf(var + 1e-5f);
  float* y = Y + (long)row * DIM;
  for (int c = threadIdx.x; c < DIM; c += 256)
    y[c] = (x[c] - mean) * inv * g[c] + b[c];
}

// ---------------- WMMA bf16 GEMM: Y = act(X @ W^T + bias) (+ Res) ----------------
// X: [M,K] f32 row-major, W: [N,K] f32 row-major, Y/Res: [M,N].
// Block = 256 threads = 8 wave32s; block tile = 32 rows x 256 cols.
// Each wave: 32x32 tile = 4 accumulators (2 m-subtiles x 2 n-subtiles):
// 2 A fragments + 2 B fragments -> 4 back-to-back v_wmma per K step.
// K and N are multiples of 32 at every call site.
template<int ACT, bool BIASF, bool RESF>
__global__ __launch_bounds__(256) void gemm_wmma(const float* __restrict__ X,
    const float* __restrict__ W, const float* __restrict__ bias,
    const float* __restrict__ Res, float* __restrict__ Y,
    int M, int N, int K) {
  __shared__ __bf16 As[32 * 32];
  const int lane = threadIdx.x & 31;
  const int wave = threadIdx.x >> 5;
  const int m0 = blockIdx.y * 32;
  int n0 = blockIdx.x * 256 + wave * 32;
  n0 = (n0 < N - 32) ? n0 : (N - 32);       // clamp: duplicate tiles write identical values
  const bool mfull = (m0 + 32 <= M);        // block-uniform

  // A fragment: lane l holds row (l&15), K rows (l>>4)*16 + e
  const int a_m  = lane & 15;
  const int a_kb = (lane >> 4) * 16;
  // B fragment: lane l holds col (l&15), K rows (l>>4)*16 + e
  const int b_kb = (lane >> 4) * 16;
  const float* wrow0 = W + (long)(n0 + (lane & 15)) * K;
  const float* wrow1 = W + (long)(n0 + 16 + (lane & 15)) * K;

  // staging coords (fast path): 256 threads x float4 = 32x32 tile
  const int st_r = threadIdx.x >> 3;
  const int st_c = (threadIdx.x & 7) * 4;

  v8f acc00 = {}, acc01 = {}, acc10 = {}, acc11 = {};
  for (int k0 = 0; k0 < K; k0 += 32) {
    __syncthreads();
    if (mfull) {
      const float4 v = *(const float4*)&X[(long)(m0 + st_r) * K + k0 + st_c];
      __bf16* dst = &As[st_r * 32 + st_c];
      dst[0] = (__bf16)v.x; dst[1] = (__bf16)v.y;
      dst[2] = (__bf16)v.z; dst[3] = (__bf16)v.w;
    } else {
      for (int i = threadIdx.x; i < 1024; i += 256) {
        const int r = i >> 5, c = i & 31;
        const int m = m0 + r;
        As[i] = (__bf16)((m < M) ? X[(long)m * K + k0 + c] : 0.f);
      }
    }
    if (k0 + 32 < K) {                       // pull next K tile toward the WGP
      if (mfull) __builtin_prefetch(&X[(long)(m0 + st_r) * K + k0 + 32 + st_c], 0, 0);
      __builtin_prefetch(&wrow0[k0 + 32 + b_kb], 0, 0);
      __builtin_prefetch(&wrow1[k0 + 32 + b_kb], 0, 0);
    }
    __syncthreads();
    v16bf a0, a1, b0, b1;
#pragma unroll
    for (int e = 0; e < 16; ++e) {
      const int k = a_kb + e;
      a0[e] = As[a_m * 32 + k];
      a1[e] = As[(a_m + 16) * 32 + k];
    }
#pragma unroll
    for (int e = 0; e < 16; ++e) {
      b0[e] = (__bf16)wrow0[k0 + b_kb + e];
      b1[e] = (__bf16)wrow1[k0 + b_kb + e];
    }
    acc00 = __builtin_amdgcn_wmma_f32_16x16x32_bf16(false, a0, false, b0,
                                                    (short)0, acc00, false, false);
    acc10 = __builtin_amdgcn_wmma_f32_16x16x32_bf16(false, a1, false, b0,
                                                    (short)0, acc10, false, false);
    acc01 = __builtin_amdgcn_wmma_f32_16x16x32_bf16(false, a0, false, b1,
                                                    (short)0, acc01, false, false);
    acc11 = __builtin_amdgcn_wmma_f32_16x16x32_bf16(false, a1, false, b1,
                                                    (short)0, acc11, false, false);
  }

  // C/D layout: VGPR r -> row r + (lane>=16 ? 8 : 0), col = lane&15
  const int mo = (lane >> 4) * 8;
  const int nA = n0 + (lane & 15);
  const int nB = nA + 16;
  const float bvA = BIASF ? bias[nA] : 0.f;
  const float bvB = BIASF ? bias[nB] : 0.f;
#pragma unroll
  for (int r = 0; r < 8; ++r) {
    const int m = m0 + mo + r;
    if (m < M) {
      float v = acc00[r] + bvA;
      if (ACT == 1) v = geluf(v);
      if (RESF) v += Res[(long)m * N + nA];
      Y[(long)m * N + nA] = v;
      float u = acc01[r] + bvB;
      if (ACT == 1) u = geluf(u);
      if (RESF) u += Res[(long)m * N + nB];
      Y[(long)m * N + nB] = u;
    }
  }
#pragma unroll
  for (int r = 0; r < 8; ++r) {
    const int m = m0 + 16 + mo + r;
    if (m < M) {
      float v = acc10[r] + bvA;
      if (ACT == 1) v = geluf(v);
      if (RESF) v += Res[(long)m * N + nA];
      Y[(long)m * N + nA] = v;
      float u = acc11[r] + bvB;
      if (ACT == 1) u = geluf(u);
      if (RESF) u += Res[(long)m * N + nB];
      Y[(long)m * N + nB] = u;
    }
  }
}

// ---------------- Attention: one block per (query i, b*h) ----------------
__global__ __launch_bounds__(256) void attn_kernel(const float* __restrict__ QKV,
    const float* __restrict__ bt, float* __restrict__ O) {
  __shared__ float q[HD];
  __shared__ float p[NPOS];
  __shared__ float red[256];
  const int i  = blockIdx.x;
  const int bh = blockIdx.y;
  const int b  = bh / NH, h = bh % NH;
  const long rowQ = (long)(b * NPOS + i) * (3 * DIM) + h * HD;
  for (int d = threadIdx.x; d < HD; d += 256) q[d] = QKV[rowQ + d];
  __syncthreads();
  const float scale = 0.121267812518166f;  // 68^-0.5
  float lmax = -1e30f;
  for (int j = threadIdx.x; j < NPOS; j += 256) {
    const float* k = QKV + (long)(b * NPOS + j) * (3 * DIM) + DIM + h * HD;
    float s = 0.f;
    for (int d = 0; d < HD; ++d) s += q[d] * k[d];
    s = s * scale + bt[h * BTW + (i - j + NPOS - 1)];
    p[j] = s;
    lmax = fmaxf(lmax, s);
  }
  red[threadIdx.x] = lmax; __syncthreads();
  for (int k = 128; k > 0; k >>= 1) { if (threadIdx.x < (unsigned)k) red[threadIdx.x] = fmaxf(red[threadIdx.x], red[threadIdx.x + k]); __syncthreads(); }
  const float mx = red[0]; __syncthreads();
  float lsum = 0.f;
  for (int j = threadIdx.x; j < NPOS; j += 256) { float e = expf(p[j] - mx); p[j] = e; lsum += e; }
  red[threadIdx.x] = lsum; __syncthreads();
  for (int k = 128; k > 0; k >>= 1) { if (threadIdx.x < (unsigned)k) red[threadIdx.x] += red[threadIdx.x + k]; __syncthreads(); }
  const float inv = 1.f / red[0];
  __syncthreads();
  for (int d = threadIdx.x; d < HD; d += 256) {
    float acc = 0.f;
    const float* v = QKV + (long)b * NPOS * (3 * DIM) + 2 * DIM + h * HD + d;
    for (int j = 0; j < NPOS; ++j) acc += p[j] * v[(long)j * (3 * DIM)];
    O[(long)(b * NPOS + i) * DIM + h * HD + d] = acc * inv;  // write in [B,N,D] layout
  }
}

// ---------------- expert scores: A[b,h,f,n] = <xu[b,f,h,:], s0[n,h,:]> ----------------
__global__ __launch_bounds__(256) void escore_kernel(const float* __restrict__ XU,
    const float* __restrict__ S0, float* __restrict__ A) {
  const int b = blockIdx.x / NPOS;
  const int f = blockIdx.x % NPOS;
  const int t = threadIdx.x;
  if (t < NH * NSEED) {
    const int h = t / NSEED, n = t % NSEED;
    const int uc = UPD / NH;  // 136
    const float* xp = XU + (long)(b * NPOS + f) * UPD + h * uc;
    const float* sp = S0 + (long)n * UPD + h * uc;
    float s = 0.f;
    for (int c = 0; c < uc; ++c) s += xp[c] * sp[c];
    A[((long)(b * NH + h) * NPOS + f) * NSEED + n] = s;
  }
}

// ---------------- softmax over the f (=NPOS) axis for each (b,h,n) ----------------
__global__ __launch_bounds__(256) void esoftmax_f(float* __restrict__ A) {
  __shared__ float red[256];
  const int n  = blockIdx.x % NSEED;
  const int bh = blockIdx.x / NSEED;
  float* base = A + (long)bh * NPOS * NSEED + n;
  const int f = threadIdx.x;
  const float v = (f < NPOS) ? base[(long)f * NSEED] : -1e30f;
  red[threadIdx.x] = v; __syncthreads();
  for (int k = 128; k > 0; k >>= 1) { if (threadIdx.x < (unsigned)k) red[threadIdx.x] = fmaxf(red[threadIdx.x], red[threadIdx.x + k]); __syncthreads(); }
  const float mx = red[0]; __syncthreads();
  const float e = (f < NPOS) ? expf(v - mx) : 0.f;
  red[threadIdx.x] = e; __syncthreads();
  for (int k = 128; k > 0; k >>= 1) { if (threadIdx.x < (unsigned)k) red[threadIdx.x] += red[threadIdx.x + k]; __syncthreads(); }
  const float inv = 1.f / red[0];
  if (f < NPOS) base[(long)f * NSEED] = e * inv;
}

// ---------------- renormalize over n (seed axis) ----------------
__global__ __launch_bounds__(32) void enorm_n(float* __restrict__ A) {
  const int b = blockIdx.x / NPOS;
  const int f = blockIdx.x % NPOS;
  const int h = threadIdx.x;
  if (h < NH) {
    float* p = A + ((long)(b * NH + h) * NPOS + f) * NSEED;
    float s = 0.f;
    for (int n = 0; n < NSEED; ++n) s += p[n];
    const float inv = 1.f / (1e-7f + s);
    for (int n = 0; n < NSEED; ++n) p[n] *= inv;
  }
}

// ---------------- mean_attn[n] = mean over (b,h,f) ----------------
__global__ __launch_bounds__(256) void emean_kernel(const float* __restrict__ A,
                                                    float* __restrict__ out) {
  __shared__ float red[256];
  const int n = blockIdx.x;
  const long T = (long)BSZ * NH * NPOS;
  float s = 0.f;
  for (long i = threadIdx.x; i < T; i += 256) s += A[i * NSEED + n];
  red[threadIdx.x] = s; __syncthreads();
  for (int k = 128; k > 0; k >>= 1) { if (threadIdx.x < (unsigned)k) red[threadIdx.x] += red[threadIdx.x + k]; __syncthreads(); }
  if (threadIdx.x == 0) out[n] = red[0] / (float)T;
}

// ---------------- o2[b,f,c'] = sum_n a[b,h,f,n] * s1[n,c']  (h = c'/uc) ----------------
__global__ __launch_bounds__(256) void eo2_kernel(const float* __restrict__ A,
    const float* __restrict__ S1, float* __restrict__ O2) {
  const int b = blockIdx.x / NPOS;
  const int f = blockIdx.x % NPOS;
  const int uc = UPD / NH;
  for (int c = threadIdx.x; c < UPD; c += 256) {
    const int h = c / uc;
    const float* ap = A + ((long)(b * NH + h) * NPOS + f) * NSEED;
    float s = 0.f;
    for (int n = 0; n < NSEED; ++n) s += ap[n] * S1[(long)n * UPD + c];
    O2[(long)(b * NPOS + f) * UPD + c] = s;
  }
}

extern "C" void kernel_launch(void* const* d_in, const int* in_sizes, int n_in,
                              void* d_out, int out_size, void* d_ws, size_t ws_size,
                              hipStream_t stream) {
  (void)in_sizes; (void)n_in; (void)out_size; (void)ws_size;
  const float* x       = (const float*)d_in[0];
  const float* seed    = (const float*)d_in[1];
  const float* ln1_g   = (const float*)d_in[2];
  const float* ln1_b   = (const float*)d_in[3];
  const float* w_qkv   = (const float*)d_in[4];
  const float* w_proj  = (const float*)d_in[5];
  const float* b_proj  = (const float*)d_in[6];
  const float* b_table = (const float*)d_in[7];
  const float* ln2_g   = (const float*)d_in[8];
  const float* ln2_b   = (const float*)d_in[9];
  const float* mlp_w1  = (const float*)d_in[10];
  const float* mlp_b1  = (const float*)d_in[11];
  const float* mlp_w2  = (const float*)d_in[12];
  const float* mlp_b2  = (const float*)d_in[13];
  const float* eln1_g  = (const float*)d_in[14];
  const float* eln1_b  = (const float*)d_in[15];
  const float* w_trans = (const float*)d_in[16];
  const float* b_trans = (const float*)d_in[17];
  const float* w0      = (const float*)d_in[18];
  const float* b0      = (const float*)d_in[19];
  const float* w1      = (const float*)d_in[20];
  const float* b1      = (const float*)d_in[21];
  const float* w_proj2 = (const float*)d_in[22];
  const float* b_proj2 = (const float*)d_in[23];
  const float* eln2_g  = (const float*)d_in[24];
  const float* eln2_b  = (const float*)d_in[25];
  const float* emlp_w1 = (const float*)d_in[26];
  const float* emlp_b1 = (const float*)d_in[27];
  const float* emlp_w2 = (const float*)d_in[28];
  const float* emlp_b2 = (const float*)d_in[29];
  float* out = (float*)d_out;

  // workspace layout (floats)
  float* w      = (float*)d_ws;
  float* buf_t  = w;                                     // M*D      (LN / attn-out temp)
  float* buf_q  = buf_t + (long)MROWS * DIM;             // M*3D     (qkv, ln temps, O2)
  float* buf_h  = buf_q + (long)MROWS * 3 * DIM;         // M*MLP1   (hidden / xu)
  float* buf_x  = buf_h + (long)MROWS * MLP1N;           // M*D      (running residual x)
  float* buf_A  = buf_x + (long)MROWS * DIM;             // B*H*N*NS
  float* buf_s0 = buf_A + (long)BSZ * NH * NPOS * NSEED; // 32*UPD (padded)
  float* buf_s1 = buf_s0 + 32 * UPD;                     // 32*UPD (padded)

  const dim3 blk(256);
  auto gg = [](int M, int N) { return dim3((unsigned)((N + 255) / 256), (unsigned)((M + 31) / 32)); };

  // ---- attention block ----
  ln_kernel<<<MROWS, blk, 0, stream>>>(x, ln1_g, ln1_b, buf_t);
  gemm_wmma<0, false, false><<<gg(MROWS, 3 * DIM), blk, 0, stream>>>(buf_t, w_qkv, nullptr, nullptr, buf_q, MROWS, 3 * DIM, DIM);
  attn_kernel<<<dim3(NPOS, BSZ * NH), blk, 0, stream>>>(buf_q, b_table, buf_t);
  gemm_wmma<0, true, true><<<gg(MROWS, DIM), blk, 0, stream>>>(buf_t, w_proj, b_proj, x, buf_x, MROWS, DIM, DIM);

  // ---- MLP block ----
  ln_kernel<<<MROWS, blk, 0, stream>>>(buf_x, ln2_g, ln2_b, buf_q);
  gemm_wmma<1, true, false><<<gg(MROWS, MLP1N), blk, 0, stream>>>(buf_q, mlp_w1, mlp_b1, nullptr, buf_h, MROWS, MLP1N, DIM);
  gemm_wmma<0, true, true><<<gg(MROWS, DIM), blk, 0, stream>>>(buf_h, mlp_w2, mlp_b2, buf_x, buf_x, MROWS, DIM, MLP1N);

  // ---- expert block ----
  ln_kernel<<<MROWS, blk, 0, stream>>>(buf_x, eln1_g, eln1_b, buf_q);
  gemm_wmma<0, true, false><<<gg(MROWS, UPD), blk, 0, stream>>>(buf_q, w_trans, b_trans, nullptr, buf_h, MROWS, UPD, DIM);
  gemm_wmma<0, true, false><<<gg(NSEED, UPD), blk, 0, stream>>>(seed, w0, b0, nullptr, buf_s0, NSEED, UPD, DIM);
  gemm_wmma<0, true, false><<<gg(NSEED, UPD), blk, 0, stream>>>(seed, w1, b1, nullptr, buf_s1, NSEED, UPD, DIM);
  escore_kernel<<<BSZ * NPOS, blk, 0, stream>>>(buf_h, buf_s0, buf_A);
  esoftmax_f<<<BSZ * NH * NSEED, blk, 0, stream>>>(buf_A);
  enorm_n<<<BSZ * NPOS, dim3(32), 0, stream>>>(buf_A);
  emean_kernel<<<NSEED, blk, 0, stream>>>(buf_A, out + (long)MROWS * DIM);
  eo2_kernel<<<BSZ * NPOS, blk, 0, stream>>>(buf_A, buf_s1, buf_q);
  gemm_wmma<0, true, true><<<gg(MROWS, DIM), blk, 0, stream>>>(buf_q, w_proj2, b_proj2, buf_x, buf_x, MROWS, DIM, UPD);

  // ---- final MLP, writes straight into d_out ----
  ln_kernel<<<MROWS, blk, 0, stream>>>(buf_x, eln2_g, eln2_b, buf_q);
  gemm_wmma<1, true, false><<<gg(MROWS, MLP2N), blk, 0, stream>>>(buf_q, emlp_w1, emlp_b1, nullptr, buf_h, MROWS, MLP2N, DIM);
  gemm_wmma<0, true, true><<<gg(MROWS, DIM), blk, 0, stream>>>(buf_h, emlp_w2, emlp_b2, buf_x, out, MROWS, DIM, MLP2N);
}
